// SimpleDecoderRetention_10376640987874
// MI455X (gfx1250) — matrix operand order repaired and checked by reading, not compiled, not run on hardware
//
#include <hip/hip_runtime.h>
#include <hip/hip_bf16.h>

typedef __bf16 bf16;
typedef __attribute__((ext_vector_type(16))) __bf16 v16bf;
typedef __attribute__((ext_vector_type(8)))  __bf16 v8bf;
typedef __attribute__((ext_vector_type(8)))  float   v8f;
typedef __attribute__((ext_vector_type(4)))  float   v4f;

#define WMMA_BF16(A, B, C) \
  __builtin_amdgcn_wmma_f32_16x16x32_bf16(false, (A), false, (B), (short)0, (C), false, false)

// hardware exp2 (v_exp_f32)
__device__ __forceinline__ float fexp2(float x) { return __builtin_amdgcn_exp2f(x); }

// B-operand (and generic contiguous-16) load: 16 contiguous bf16 (two b128s)
__device__ __forceinline__ v16bf ld_b16(const bf16* p) {
  union { v16bf v; v8bf h[2]; } u;
  u.h[0] = *(const v8bf*)(p);
  u.h[1] = *(const v8bf*)(p + 8);
  return u.v;
}
// A-operand load: elements {o8..o8+7} U {o8+16..o8+23} of a 32-wide K chunk
__device__ __forceinline__ v16bf ld_a16(const bf16* p, int o8) {
  union { v16bf v; v8bf h[2]; } u;
  u.h[0] = *(const v8bf*)(p + o8);
  u.h[1] = *(const v8bf*)(p + o8 + 16);
  return u.v;
}

// ---------------------------------------------------------------------------
// Transpose + convert W [512][64] f32  ->  Wt [64][512] bf16
// ---------------------------------------------------------------------------
__global__ void __launch_bounds__(256)
wt_kernel(const float* __restrict__ W, bf16* __restrict__ Wt) {
  int idx = blockIdx.x * 256 + threadIdx.x;   // 0..32767
  int k  = idx & 511;
  int cc = idx >> 9;
  Wt[(size_t)cc * 512 + k] = (bf16)W[(size_t)k * 64 + cc];
}

// ---------------------------------------------------------------------------
// WMMA projection: Out = xpos(X @ W) as bf16.
// Each wave: one 16(rows) x 16(cols) output tile, K-loop over 512 (16 x 32).
// xposMode: 0 = none (V), 1 = normal (Q), 2 = downscale (K)
// transOut: 0 = row-major [row][64], 1 = transposed [b][64][seqlen]
// ---------------------------------------------------------------------------
__global__ void __launch_bounds__(256)
proj_xpos_kernel(const float* __restrict__ Xsrc, const bf16* __restrict__ Wt,
                 bf16* __restrict__ Out, int seqlen, int xposMode, int transOut) {
  const int lane = threadIdx.x & 31;
  const int wv   = threadIdx.x >> 5;
  const int gw   = blockIdx.x * 8 + wv;
  const int rt   = gw >> 2;          // row tile
  const int ct   = gw & 3;           // col tile (64 = 4 x 16)
  const int col  = lane & 15;
  const int hi   = lane >> 4;
  const int o8   = hi * 8, o16 = hi * 16;

  const int rowA = rt * 16 + col;    // A-operand row (token)
  const int c    = ct * 16 + col;    // output dim for this lane (B column)
  const float* Xr = Xsrc + (size_t)rowA * 512;
  const bf16*  Wr = Wt   + (size_t)c    * 512;

  v8f acc = {};
#pragma unroll 4
  for (int kk = 0; kk < 16; ++kk) {
    const int o = kk * 32;
    union { v16bf v; v8bf h[2]; } ua;
    v4f x0 = *(const v4f*)(Xr + o + o8);
    v4f x1 = *(const v4f*)(Xr + o + o8 + 4);
    v4f x2 = *(const v4f*)(Xr + o + o8 + 16);
    v4f x3 = *(const v4f*)(Xr + o + o8 + 20);
#pragma unroll
    for (int e = 0; e < 4; ++e) {
      ua.h[0][e]     = (bf16)x0[e];
      ua.h[0][e + 4] = (bf16)x1[e];
      ua.h[1][e]     = (bf16)x2[e];
      ua.h[1][e + 4] = (bf16)x3[e];
    }
    v16bf bv = ld_a16(Wr + o, o8);
    acc = WMMA_BF16(ua.v, bv, acc);
  }

  const int rowBase = rt * 16;
  const int bidx = rowBase / seqlen;

  if (xposMode) {
    // XPOS rotary: lane holds fixed output dim d; pair partner lives in lane^1.
    const int d = c, j = d >> 1;
    const float invf = fexp2(-(float)j * (13.28771237954945f / 32.f)); // 10000^(-j/32)
    float lb = __log2f((2.f * (float)j + 25.6f) * (1.f / 89.6f));      // log2(base_j)
    if (xposMode == 2) lb = -lb;                                        // downscale
#pragma unroll
    for (int r = 0; r < 8; ++r) {
      float cv = acc[r];
      float nb = __shfl_xor(cv, 1, 32);           // value at dim d^1, same token
      int n = (rowBase + r + 8 * hi) % seqlen;    // position within sequence
      float ang = (float)n * invf;
      float sv, co;
      __sincosf(ang, &sv, &co);
      float scale = fexp2((float)n * (1.f / 512.f) * lb);  // base^(n/512)
      float rot = (d & 1) ? nb : -nb;             // rotate_every_two
      acc[r] = (cv * co + rot * sv) * scale;
    }
  }

  if (!transOut) {
#pragma unroll
    for (int r = 0; r < 8; ++r)
      Out[(size_t)(rowBase + r + 8 * hi) * 64 + c] = (bf16)acc[r];
  } else {
    // Out layout [b][64][seqlen]; lane's 8 rows are consecutive m -> one 16B store
    v8bf pk;
#pragma unroll
    for (int r = 0; r < 8; ++r) pk[r] = (bf16)acc[r];
    const int m0 = (rowBase % seqlen) + 8 * hi;
    *(v8bf*)(Out + ((size_t)bidx * 64 + c) * seqlen + m0) = pk;
  }
}

// ---------------------------------------------------------------------------
// Retention: out[b, n0:n0+16, 0:64] = sum_m ((Q K^T) * gamma^(n-m) [n>=m]) V
// One wave per 16-row n-tile; m chunked by 32 (only m <= n0+15 -> exact
// causality, skips the entire dead half of Mem).
// ---------------------------------------------------------------------------
__global__ void __launch_bounds__(256)
retention_kernel(const bf16* __restrict__ Qb, const bf16* __restrict__ Kb,
                 const bf16* __restrict__ Vt, float* __restrict__ Outp,
                 int N, int M) {
  __shared__ bf16 lds[8 * 16 * 40];   // per-wave 16x32 S tile, stride 40 (pad)
  const int lane = threadIdx.x & 31;
  const int wv   = threadIdx.x >> 5;
  const int gw   = blockIdx.x * 8 + wv;
  const int ntiles = N >> 4;
  const int b  = gw / ntiles;
  const int t  = gw % ntiles;
  const int n0 = t << 4;
  const int col = lane & 15;
  const int hi  = lane >> 4;
  const int o8  = hi * 8, o16 = hi * 16;

  bf16* S = lds + wv * (16 * 40);

  const bf16* Qrow = Qb + ((size_t)b * N + n0 + col) * 64;
  const v16bf qa0 = ld_a16(Qrow, o8);        // d 0..31 chunk (A layout)
  const v16bf qa1 = ld_a16(Qrow + 32, o8);   // d 32..63 chunk

  v8f acc0 = {}, acc1 = {}, acc2 = {}, acc3 = {};

  const float LOG2G = -0.04580369f;          // log2(0.96875)
  float g8[8];
  g8[0] = 1.f;
#pragma unroll
  for (int r = 1; r < 8; ++r) g8[r] = g8[r - 1] * 0.96875f;

  const bf16* Kbase = Kb + (size_t)b * M * 64;
  const bf16* Vbase = Vt + (size_t)b * 64 * M;

  for (int m0 = 0; m0 <= n0 + 15; m0 += 32) {
#pragma unroll
    for (int mt = 0; mt < 2; ++mt) {
      const int mb = m0 + mt * 16;
      const bf16* Krow = Kbase + (size_t)(mb + col) * 64;   // B-op: K row = col of K^T
      v16bf kb0 = ld_a16(Krow, o8);
      v16bf kb1 = ld_a16(Krow + 32, o8);
      v8f s = {};
      s = WMMA_BF16(qa0, kb0, s);
      s = WMMA_BF16(qa1, kb1, s);
      // decay: n-m = Cc + r per lane; one exp2 + table of gamma^r
      const int Cc = n0 + 8 * hi - (mb + col);
      const float lanef = fexp2((float)Cc * LOG2G);
#pragma unroll
      for (int r = 0; r < 8; ++r) {
        float dly = (Cc + r >= 0) ? lanef * g8[r] : 0.f;
        float sv  = s[r] * dly;
        S[(r + 8 * hi) * 40 + mt * 16 + col] = (bf16)sv;    // C/D -> row-major LDS
      }
    }

    // Re-read S in A layout (lane row = col, K(m) = {o16..o16+15})
    union { v16bf v; v8bf h[2]; } us;
    us.h[0] = *(const v8bf*)(S + col * 40 + o16);
    us.h[1] = *(const v8bf*)(S + col * 40 + o16 + 8);

    const bf16* Vk = Vbase + m0 + o16;   // Vt[b][v][m]: contiguous m per lane
    acc0 = WMMA_BF16(us.v, ld_b16(Vk + (size_t)(0 * 16 + col) * M), acc0);
    acc1 = WMMA_BF16(us.v, ld_b16(Vk + (size_t)(1 * 16 + col) * M), acc1);
    acc2 = WMMA_BF16(us.v, ld_b16(Vk + (size_t)(2 * 16 + col) * M), acc2);
    acc3 = WMMA_BF16(us.v, ld_b16(Vk + (size_t)(3 * 16 + col) * M), acc3);

    __builtin_prefetch(Kbase + (size_t)(m0 + 32 + col) * 64, 0, 3);
  }

  float* Or = Outp + ((size_t)b * N + n0 + 8 * hi) * 64 + col;
#pragma unroll
  for (int r = 0; r < 8; ++r) {
    Or[(size_t)r * 64 + 0]  = acc0[r];
    Or[(size_t)r * 64 + 16] = acc1[r];
    Or[(size_t)r * 64 + 32] = acc2[r];
    Or[(size_t)r * 64 + 48] = acc3[r];
  }
}

// ---------------------------------------------------------------------------
extern "C" void kernel_launch(void* const* d_in, const int* in_sizes, int n_in,
                              void* d_out, int out_size, void* d_ws, size_t ws_size,
                              hipStream_t stream) {
  (void)in_sizes; (void)n_in; (void)out_size; (void)ws_size;
  const float* X   = (const float*)d_in[0];   // [4,2048,512]
  const float* Mem = (const float*)d_in[1];   // [4,4096,512]
  const float* WQ  = (const float*)d_in[2];   // [512,64]
  const float* WK  = (const float*)d_in[3];
  const float* WV  = (const float*)d_in[4];
  float* out = (float*)d_out;                 // [4,2048,64]

  // workspace layout (bf16): 3 x Wt[64,512], Q[4,2048,64], K[4,4096,64], Vt[4,64,4096]
  bf16* Wqt = (bf16*)d_ws;
  bf16* Wkt = Wqt + 64 * 512;
  bf16* Wvt = Wkt + 64 * 512;
  bf16* Qb  = Wvt + 64 * 512;
  bf16* Kb  = Qb + (size_t)4 * 2048 * 64;
  bf16* Vtb = Kb + (size_t)4 * 4096 * 64;

  wt_kernel<<<128, 256, 0, stream>>>(WQ, Wqt);
  wt_kernel<<<128, 256, 0, stream>>>(WK, Wkt);
  wt_kernel<<<128, 256, 0, stream>>>(WV, Wvt);

  // Q: 8192 rows -> 512 row tiles * 4 col tiles = 2048 waves = 256 blocks
  proj_xpos_kernel<<<256, 256, 0, stream>>>(X,   Wqt, Qb,  2048, 1, 0);
  // K: 16384 rows -> 4096 waves = 512 blocks (xpos downscale)
  proj_xpos_kernel<<<512, 256, 0, stream>>>(Mem, Wkt, Kb,  4096, 2, 0);
  // V: no xpos, transposed output
  proj_xpos_kernel<<<512, 256, 0, stream>>>(Mem, Wvt, Vtb, 4096, 0, 1);

  // 4 batches * 128 n-tiles = 512 waves = 64 blocks of 8 waves
  retention_kernel<<<64, 256, 0, stream>>>(Qb, Kb, Vtb, out, 2048, 4096);
}
